// FeedForward_50002009260705
// MI455X (gfx1250) — matrix operand order repaired and checked by reading, not compiled, not run on hardware
//
#include <hip/hip_runtime.h>
#include <hip/hip_bf16.h>

// Problem constants (match reference): B=8, S=4096, D=1024, F=2048, E=8, K=2
#define BB 8
#define SS 4096
#define DD 1024
#define FF 2048
#define EE 8
#define KK 2
#define TT (BB * SS)            // 32768 tokens
#define TILE_M 32               // tokens per workgroup tile (2 row-blocks of 16)
#define NTILES (TT / TILE_M)    // 1024 tile slots per expert
#define FC 256                  // F-chunk for fused GEMM1->GEMM2
#define NCHUNK (FF / FC)        // 8
#define XS (DD + 8)             // padded LDS row stride for x tile (bf16 elems)
#define HS (FC + 8)             // padded LDS row stride for h chunk (bf16 elems)

typedef __attribute__((ext_vector_type(16))) __bf16 v16bf;
typedef __attribute__((ext_vector_type(8)))  float  v8f;

union Frag16 { v16bf v; uint4 q[2]; };

__device__ __forceinline__ unsigned int f32_to_bf16_rne(float f) {
    unsigned int u = __float_as_uint(f);
    unsigned int r = u + 0x7FFFu + ((u >> 16) & 1u);   // round-to-nearest-even
    return r >> 16;
}

// silu via native reciprocal: avoids the IEEE divide expansion
// (v_div_scale/v_div_fixup ladder); 1-ulp rcp error is below bf16 rounding.
__device__ __forceinline__ float silu_fast(float h) {
    return h * __builtin_amdgcn_rcpf(1.0f + __expf(-h));
}

// ---------------------------------------------------------------------------
// Kernel 1: zero the f32 output (atomic-accumulate target) + expert counters.
// Grid sized so each thread stores exactly one float4.
// ---------------------------------------------------------------------------
__global__ void moe_init_kernel(float* __restrict__ out, int* __restrict__ counts) {
    size_t i = (size_t)blockIdx.x * blockDim.x + threadIdx.x;
    float4 z = {0.f, 0.f, 0.f, 0.f};
    ((float4*)out)[i] = z;
    if (blockIdx.x == 0 && threadIdx.x < EE) counts[threadIdx.x] = 0;
}

// ---------------------------------------------------------------------------
// Kernel 2: routing — build per-expert (token, gate) lists via atomic append.
// ---------------------------------------------------------------------------
__global__ void moe_route_kernel(const int* __restrict__ idx,
                                 const float* __restrict__ probs,
                                 int* __restrict__ counts,
                                 int* __restrict__ toks,
                                 float* __restrict__ gates) {
    int t = blockIdx.x * blockDim.x + threadIdx.x;
    if (t >= TT) return;
#pragma unroll
    for (int k = 0; k < KK; ++k) {
        int   e = idx[t * KK + k];
        float g = probs[t * KK + k];
        int slot = atomicAdd(&counts[e], 1);
        toks[e * TT + slot]  = t;
        gates[e * TT + slot] = g;
    }
}

// ---------------------------------------------------------------------------
// Kernel 3: f32 -> bf16 weight conversion (4 elems / thread).
// ---------------------------------------------------------------------------
__global__ void moe_cvt_bf16_kernel(const float* __restrict__ src,
                                    unsigned short* __restrict__ dst, int n4) {
    int i = blockIdx.x * blockDim.x + threadIdx.x;
    if (i >= n4) return;
    float4 f = ((const float4*)src)[i];
    ushort4 o;
    o.x = (unsigned short)f32_to_bf16_rne(f.x);
    o.y = (unsigned short)f32_to_bf16_rne(f.y);
    o.z = (unsigned short)f32_to_bf16_rne(f.z);
    o.w = (unsigned short)f32_to_bf16_rne(f.w);
    ((ushort4*)dst)[i] = o;
}

// ---------------------------------------------------------------------------
// Kernel 4: fused grouped MoE FFN tile kernel.
// One workgroup = 8 wave32 = one 32-token tile (2 row-blocks) of one expert.
// Each B fragment feeds 2 WMMAs (row-blocks), halving L2 weight traffic vs
// a 16-token tile.  For each F-chunk: GEMM1 (x@W1^T -> silu -> LDS bf16),
// GEMM2 (h@W2^T into persistent 32x1024 f32 accumulators).  Finally
// gate-scale + f32 atomic combine (each token has K=2 contributions).
// ---------------------------------------------------------------------------
__launch_bounds__(256)
__global__ void moe_ffn_kernel(const float* __restrict__ x,
                               const unsigned short* __restrict__ w1b,
                               const unsigned short* __restrict__ w2b,
                               const int* __restrict__ counts,
                               const int* __restrict__ toks,
                               const float* __restrict__ gates,
                               float* __restrict__ out) {
    __shared__ __align__(16) unsigned short x_lds[TILE_M * XS];   // 66 KB
    __shared__ __align__(16) unsigned short h_lds[TILE_M * HS];   // 16.9 KB
    __shared__ int   tok_s[TILE_M];
    __shared__ float gate_s[TILE_M];

    const int e    = blockIdx.x / NTILES;
    const int tile = blockIdx.x % NTILES;
    const int cnt  = counts[e];
    const int row0 = tile * TILE_M;
    if (row0 >= cnt) return;             // uniform early-exit for empty tiles

    const int tid  = threadIdx.x;
    const int wave = tid >> 5;
    const int lane = tid & 31;
    const int ln   = lane & 15;          // column / row index within 16
    const int lh   = lane >> 4;          // half-wave select

    if (tid < TILE_M) {
        int r = row0 + tid;
        if (r < cnt) { tok_s[tid] = toks[e * TT + r]; gate_s[tid] = gates[e * TT + r]; }
        else         { tok_s[tid] = 0;                gate_s[tid] = 0.0f; }
    }
    __syncthreads();

    // Gather x rows -> LDS (f32 -> bf16), packed 8-byte LDS stores.
    {
        const int r  = tid >> 3;             // 0..31
        const int c0 = (tid & 7) * 128;
        const float* xr = x + (size_t)tok_s[r] * DD;
#pragma unroll 4
        for (int c = 0; c < 128; c += 4) {
            float4 f = *(const float4*)(xr + c0 + c);
            uint2 p;
            p.x = f32_to_bf16_rne(f.x) | (f32_to_bf16_rne(f.y) << 16);
            p.y = f32_to_bf16_rne(f.z) | (f32_to_bf16_rne(f.w) << 16);
            *(uint2*)&x_lds[r * XS + c0 + c] = p;
        }
    }
    __syncthreads();

    // Persistent output accumulators: 8 col-blocks x 2 row-blocks (32x128/wave)
    v8f oacc[8][2];
#pragma unroll
    for (int j = 0; j < 8; ++j) {
        oacc[j][0] = (v8f){0.f,0.f,0.f,0.f,0.f,0.f,0.f,0.f};
        oacc[j][1] = (v8f){0.f,0.f,0.f,0.f,0.f,0.f,0.f,0.f};
    }

    for (int ch = 0; ch < NCHUNK; ++ch) {
        // ---- GEMM1: 2 col-blocks x 2 row-blocks of silu(x @ W1^T) per wave
        v8f hacc[2][2];
#pragma unroll
        for (int t = 0; t < 2; ++t) {
            hacc[t][0] = (v8f){0.f,0.f,0.f,0.f,0.f,0.f,0.f,0.f};
            hacc[t][1] = (v8f){0.f,0.f,0.f,0.f,0.f,0.f,0.f,0.f};
        }
        const int fBase = ch * FC + wave * 32;
        const unsigned short* w1r0 = w1b + ((size_t)e * FF + fBase +      ln) * DD + lh * 8;
        const unsigned short* w1r1 = w1b + ((size_t)e * FF + fBase + 16 + ln) * DD + lh * 8;
#pragma unroll 2
        for (int k0 = 0; k0 < DD; k0 += 32) {
            Frag16 a0, a1;
            a0.q[0] = *(const uint4*)&x_lds[ ln       * XS + k0 +      lh * 8];
            a0.q[1] = *(const uint4*)&x_lds[ ln       * XS + k0 + 16 + lh * 8];
            a1.q[0] = *(const uint4*)&x_lds[(ln + 16) * XS + k0 +      lh * 8];
            a1.q[1] = *(const uint4*)&x_lds[(ln + 16) * XS + k0 + 16 + lh * 8];
            Frag16 b0, b1;
            const uint4* pb0 = (const uint4*)(w1r0 + k0);
            const uint4* pb1 = (const uint4*)(w1r1 + k0);
            b0.q[0] = pb0[0]; b0.q[1] = pb0[2];
            b1.q[0] = pb1[0]; b1.q[1] = pb1[2];
            hacc[0][0] = __builtin_amdgcn_wmma_f32_16x16x32_bf16(false, a0.v, false, b0.v,
                                                                 (short)0, hacc[0][0], false, false);
            hacc[0][1] = __builtin_amdgcn_wmma_f32_16x16x32_bf16(false, a1.v, false, b0.v,
                                                                 (short)0, hacc[0][1], false, false);
            hacc[1][0] = __builtin_amdgcn_wmma_f32_16x16x32_bf16(false, a0.v, false, b1.v,
                                                                 (short)0, hacc[1][0], false, false);
            hacc[1][1] = __builtin_amdgcn_wmma_f32_16x16x32_bf16(false, a1.v, false, b1.v,
                                                                 (short)0, hacc[1][1], false, false);
        }

        // All waves must be done reading h_lds (previous chunk's GEMM2).
        __syncthreads();
        {
            const int cb0 = wave * 32;
#pragma unroll
            for (int t = 0; t < 2; ++t) {
#pragma unroll
                for (int rb = 0; rb < 2; ++rb) {
#pragma unroll
                    for (int v = 0; v < 8; ++v) {
                        const int m = v + 8 * lh + 16 * rb;
                        const float h = silu_fast(hacc[t][rb][v]);
                        h_lds[m * HS + cb0 + t * 16 + ln] =
                            (unsigned short)f32_to_bf16_rne(h);
                    }
                }
            }
        }
        __syncthreads();

        // ---- GEMM2: oacc += h_chunk @ W2_chunk^T (8 col-blocks x 2 row-blocks)
#pragma unroll
        for (int k0 = 0; k0 < FC; k0 += 32) {
            Frag16 a0, a1;
            a0.q[0] = *(const uint4*)&h_lds[ ln       * HS + k0 +      lh * 8];
            a0.q[1] = *(const uint4*)&h_lds[ ln       * HS + k0 + 16 + lh * 8];
            a1.q[0] = *(const uint4*)&h_lds[(ln + 16) * HS + k0 +      lh * 8];
            a1.q[1] = *(const uint4*)&h_lds[(ln + 16) * HS + k0 + 16 + lh * 8];
            const int fk = ch * FC + k0 + lh * 8;
#pragma unroll
            for (int j = 0; j < 8; ++j) {
                const int oB = wave * 128 + j * 16;
                const uint4* pb = (const uint4*)(w2b + ((size_t)e * DD + oB + ln) * FF + fk);
                Frag16 b; b.q[0] = pb[0]; b.q[1] = pb[2];
                oacc[j][0] = __builtin_amdgcn_wmma_f32_16x16x32_bf16(false, a0.v, false, b.v,
                                                                     (short)0, oacc[j][0], false, false);
                oacc[j][1] = __builtin_amdgcn_wmma_f32_16x16x32_bf16(false, a1.v, false, b.v,
                                                                     (short)0, oacc[j][1], false, false);
            }
        }
    }

    // ---- Gate-scale and combine (each token has K=2 contributions).
#pragma unroll
    for (int j = 0; j < 8; ++j) {
        const int oB = wave * 128 + j * 16;
#pragma unroll
        for (int rb = 0; rb < 2; ++rb) {
#pragma unroll
            for (int v = 0; v < 8; ++v) {
                const int m = v + 8 * lh + 16 * rb;
                const float val = oacc[j][rb][v] * gate_s[m];
                unsafeAtomicAdd(&out[(size_t)tok_s[m] * DD + oB + ln], val);
            }
        }
    }
}

// ---------------------------------------------------------------------------
// Host launcher.  Workspace layout (~69 MB):
//   [0]                  w1 bf16   E*F*D*2 = 33.5 MB
//   [+33.5MB]            w2 bf16   E*D*F*2 = 33.5 MB
//   [+67.1MB]            counts    256 B
//   [+...]               token ids E*T*4   = 1 MB
//   [+...]               gates     E*T*4   = 1 MB
// ---------------------------------------------------------------------------
extern "C" void kernel_launch(void* const* d_in, const int* in_sizes, int n_in,
                              void* d_out, int out_size, void* d_ws, size_t ws_size,
                              hipStream_t stream) {
    const float* x     = (const float*)d_in[0];
    const float* probs = (const float*)d_in[1];
    const int*   idx   = (const int*)d_in[2];
    const float* w1    = (const float*)d_in[3];
    const float* w2    = (const float*)d_in[4];
    float* out = (float*)d_out;

    char* ws = (char*)d_ws;
    size_t off = 0;
    unsigned short* w1b = (unsigned short*)(ws + off); off += (size_t)EE * FF * DD * 2;
    unsigned short* w2b = (unsigned short*)(ws + off); off += (size_t)EE * FF * DD * 2;
    int*   counts = (int*)(ws + off);   off += 256;
    int*   toks   = (int*)(ws + off);   off += (size_t)EE * TT * 4;
    float* gates  = (float*)(ws + off); off += (size_t)EE * TT * 4;

    // 1) zero output + counters (exactly TT*DD floats, one float4 per thread)
    {
        const int nblk = (TT * DD / 4) / 256;    // 32768 blocks
        moe_init_kernel<<<nblk, 256, 0, stream>>>(out, counts);
    }
    // 2) routing
    moe_route_kernel<<<TT / 256, 256, 0, stream>>>(idx, probs, counts, toks, gates);
    // 3) weight conversion to bf16
    {
        const int n4 = EE * FF * DD / 4;         // 4,194,304 float4s each
        moe_cvt_bf16_kernel<<<(n4 + 255) / 256, 256, 0, stream>>>(w1, w1b, n4);
        moe_cvt_bf16_kernel<<<(n4 + 255) / 256, 256, 0, stream>>>(w2, w2b, n4);
    }
    // 4) fused grouped FFN
    moe_ffn_kernel<<<EE * NTILES, 256, 0, stream>>>(x, w1b, w2b, counts, toks, gates, out);
}
